// Deceiver_8641474199939
// MI455X (gfx1250) — matrix operand order repaired and checked by reading, not compiled, not run on hardware
//
#include <hip/hip_runtime.h>
#include <math.h>

// ---------------------------------------------------------------------------
// CDNA5 (gfx1250) Perceiver-decoder forward pass.
// All dense math goes through v_wmma_f32_16x16x32_f16 (f16 in, f32 accum).
// LDS tiles are stored in WMMA *fragment order* so each fragment load is a
// single 32-byte vector read (2x ds_load_b128).  Global tile loads are
// float4 (global_load_b128) with branchless clamp+select bounds handling.
// ---------------------------------------------------------------------------

typedef _Float16 half_t;
typedef _Float16 v16h __attribute__((ext_vector_type(16)));
typedef _Float16 v4h  __attribute__((ext_vector_type(4)));
typedef float    v8f  __attribute__((ext_vector_type(8)));

#define NB   8       // batch
#define HW_  16384   // H*W
#define NLAT 256
#define DLAT 512
#define QDIM 29

__device__ inline v8f wmma_f16(v16h a, v16h b, v8f c) {
  // 8 args: (neg_a, A, neg_b, B, c_mod, C, reuse_a, reuse_b)
  return __builtin_amdgcn_wmma_f32_16x16x32_f16(false, a, false, b, (short)0, c,
                                                false, false);
}

// Fragment-order swizzle offsets (ISA 7.12.2, 16-bit A 16x32 / B 32x16):
//  A element (m in [0,16), k in [0,32)):
//    lane = m + 16*((k>>3)&1), half = (k&7) + 8*(k>>4)
//  B element (k in [0,32), n in [0,16)):
//    lane = n + 16*((k>>3)&1), half = (k&7) + 8*(k>>4)
// Each fragment occupies 512 halfs = lane*16 + half (32B per lane, aligned).
__device__ inline int a_swz(int m, int k) {
  return (m + 16 * ((k >> 3) & 1)) * 16 + (k & 7) + 8 * (k >> 4);
}
__device__ inline int b_swz(int k, int n) {
  return (n + 16 * ((k >> 3) & 1)) * 16 + (k & 7) + 8 * (k >> 4);
}

__device__ inline v4h cvt4(float4 v) {
  v4h h;
  h[0] = (half_t)v.x; h[1] = (half_t)v.y; h[2] = (half_t)v.z; h[3] = (half_t)v.w;
  return h;
}

__device__ inline float half_max16(float v) {
#pragma unroll
  for (int off = 1; off < 16; off <<= 1) v = fmaxf(v, __shfl_xor(v, off));
  return v;
}
__device__ inline float half_sum16(float v) {
#pragma unroll
  for (int off = 1; off < 16; off <<= 1) v += __shfl_xor(v, off);
  return v;
}

// ---- tile loaders (128 threads) -------------------------------------------
// A tile: 64 rows x 32 k from Ap[row*K + k0+..] -> fragment-order As[4*512]
__device__ inline void load_a_tile(half_t* As, const float* __restrict__ Ap,
                                   long a0, int rowBase, int k0, int K,
                                   bool vec, int tid) {
  if (vec) {  // K % 4 == 0: float4 loads, 4 contiguous halfs per store
    for (int i = tid; i < 512; i += 128) {
      int rr = i >> 3, c4 = (i & 7) << 2;
      int k = k0 + c4;
      int kc = min(k, K - 4);
      float4 v = *(const float4*)(Ap + a0 + (long)(rowBase + rr) * K + kc);
      bool ok = k < K;
      v.x = ok ? v.x : 0.f; v.y = ok ? v.y : 0.f;
      v.z = ok ? v.z : 0.f; v.w = ok ? v.w : 0.f;
      *(v4h*)(As + (rr >> 4) * 512 + a_swz(rr & 15, c4)) = cvt4(v);
    }
  } else {
    for (int i = tid; i < 64 * 32; i += 128) {
      int rr = i >> 5, cc = i & 31, k = k0 + cc;
      int kc = min(k, K - 1);
      float v = Ap[a0 + (long)(rowBase + rr) * K + kc];
      v = (k < K) ? v : 0.f;
      As[(rr >> 4) * 512 + a_swz(rr & 15, cc)] = (half_t)v;
    }
  }
}

// B tile: 32 k x 64 n from W[k*ldw + nOff + n] -> fragment-order Bs[4*512]
__device__ inline void load_b_tile(half_t* Bs, const float* __restrict__ W,
                                   int ldw, int nOff, int colBase, int k0,
                                   int K, int N, bool vec, int tid) {
  if (vec) {  // N % 4 == 0
    for (int i = tid; i < 512; i += 128) {
      int rr = i >> 4, c4 = (i & 15) << 2;
      int k = k0 + rr, n = colBase + c4;
      int kc = min(k, K - 1), nc = min(n, N - 4);
      float4 v = *(const float4*)(W + (long)kc * ldw + nOff + nc);
      bool ok = (k < K) && (n < N);
      v.x = ok ? v.x : 0.f; v.y = ok ? v.y : 0.f;
      v.z = ok ? v.z : 0.f; v.w = ok ? v.w : 0.f;
      int t = c4 >> 4, n16 = c4 & 15;
      Bs[t * 512 + b_swz(rr, n16 + 0)] = (half_t)v.x;
      Bs[t * 512 + b_swz(rr, n16 + 1)] = (half_t)v.y;
      Bs[t * 512 + b_swz(rr, n16 + 2)] = (half_t)v.z;
      Bs[t * 512 + b_swz(rr, n16 + 3)] = (half_t)v.w;
    }
  } else {
    for (int i = tid; i < 32 * 64; i += 128) {
      int rr = i >> 6, cc = i & 63;
      int k = k0 + rr, n = colBase + cc;
      int kc = min(k, K - 1), nc = min(n, N - 1);
      float v = W[(long)kc * ldw + nOff + nc];
      v = (k < K && n < N) ? v : 0.f;
      Bs[(cc >> 4) * 512 + b_swz(rr, cc & 15)] = (half_t)v;
    }
  }
}

// ---------------------------------------------------------------------------
// lat = (x @ W_l2l + b_l2l).reshape(8,256,512) + latents   (HBM-bound, 268MB)
// ---------------------------------------------------------------------------
__global__ __launch_bounds__(256) void k_l2l(const float* __restrict__ x,
                                             const float* __restrict__ W,
                                             const float* __restrict__ bias,
                                             const float* __restrict__ latents,
                                             float* __restrict__ lat) {
  __shared__ float xs[NB * DLAT];
  int tid = threadIdx.x;
  for (int i = tid; i < NB * DLAT; i += 256) xs[i] = x[i];
  __syncthreads();
  size_t j = (size_t)blockIdx.x * 256 + tid;  // 0 .. 131071
  float acc[NB];
#pragma unroll
  for (int b = 0; b < NB; ++b) acc[b] = 0.f;
  for (int k = 0; k < DLAT; ++k) {
    float w = W[(size_t)k * (NLAT * DLAT) + j];
#pragma unroll
    for (int b = 0; b < NB; ++b) acc[b] += xs[b * DLAT + k] * w;
  }
  float add = bias[j] + latents[j];
#pragma unroll
  for (int b = 0; b < NB; ++b)
    lat[(size_t)b * (NLAT * DLAT) + j] = acc[b] + add;
}

// ---------------------------------------------------------------------------
// data init: channels 0-2 zero, then Fourier encoding (same for all batches)
// ---------------------------------------------------------------------------
__global__ __launch_bounds__(256) void k_init_data(float* __restrict__ data) {
  int p = blockIdx.x * 256 + threadIdx.x;
  if (p >= HW_) return;
  int h = p >> 7, w = p & 127;
  float pos[2] = {-1.f + 2.f * h / 127.f, -1.f + 2.f * w / 127.f};
  const float PI_ = 3.14159265358979323846f;
  const float step = (log2f(5.0f) - 1.0f) / 5.0f;  // logspace(1, log2(5), 6)
  float enc[26];
#pragma unroll
  for (int d = 0; d < 2; ++d) {
#pragma unroll
    for (int i = 0; i < 6; ++i) {
      float s = exp2f(1.0f + step * i) * PI_ * pos[d];
      enc[d * 13 + i]     = sinf(s);
      enc[d * 13 + 6 + i] = cosf(s);
    }
    enc[d * 13 + 12] = pos[d];
  }
  for (int b = 0; b < NB; ++b) {
    float* o = data + ((size_t)b * HW_ + p) * QDIM;
    o[0] = 0.f; o[1] = 0.f; o[2] = 0.f;
#pragma unroll
    for (int j = 0; j < 26; ++j) o[3 + j] = enc[j];
  }
}

// ---------------------------------------------------------------------------
// LayerNorm: one wave per row (wave32), shuffle reduction
// ---------------------------------------------------------------------------
__global__ __launch_bounds__(256) void k_layernorm(const float* __restrict__ in,
                                                   float* __restrict__ out,
                                                   const float* __restrict__ g,
                                                   const float* __restrict__ b,
                                                   int D) {
  int lane = threadIdx.x & 31;
  int wave = threadIdx.x >> 5;
  long row = (long)blockIdx.x * 8 + wave;
  const float* r = in + row * D;
  float s = 0.f, ss = 0.f;
  for (int d = lane; d < D; d += 32) {
    float v = r[d];
    s += v; ss += v * v;
  }
#pragma unroll
  for (int off = 16; off; off >>= 1) {
    s  += __shfl_xor(s, off);
    ss += __shfl_xor(ss, off);
  }
  float mean = s / D;
  float var  = ss / D - mean * mean;
  float inv  = rsqrtf(var + 1e-5f);
  float* o = out + row * D;
  for (int d = lane; d < D; d += 32) o[d] = (r[d] - mean) * inv * g[d] + b[d];
}

// ---------------------------------------------------------------------------
// Generic WMMA GEMM: C[z] = A[z] @ W (+bias) (+res).  Tile 64x64, 4 waves.
// ---------------------------------------------------------------------------
__global__ __launch_bounds__(128) void k_gemm(const float* __restrict__ A,
                                              const float* __restrict__ W,
                                              const float* __restrict__ bias,
                                              const float* __restrict__ res,
                                              float* __restrict__ C,
                                              int M, int N, int K,
                                              long batchA, long batchC) {
  __shared__ __align__(32) half_t As[4 * 512];
  __shared__ __align__(32) half_t Bs[4 * 512];
  int tid = threadIdx.x, lane = tid & 31, wave = tid >> 5;
  long a0 = (long)blockIdx.z * batchA;
  long c0 = (long)blockIdx.z * batchC;
  int rowBase = blockIdx.y * 64;
  int colBase = blockIdx.x * 64;
  bool aVec = (K & 3) == 0;
  bool bVec = (N & 3) == 0;
  v8f acc[4];
#pragma unroll
  for (int t = 0; t < 4; ++t)
#pragma unroll
    for (int r = 0; r < 8; ++r) acc[t][r] = 0.f;

  for (int k0 = 0; k0 < K; k0 += 32) {
    __syncthreads();
    load_a_tile(As, A, a0, rowBase, k0, K, aVec, tid);
    load_b_tile(Bs, W, N, 0, colBase, k0, K, N, bVec, tid);
    __syncthreads();
    v16h af = *(const v16h*)(As + wave * 512 + lane * 16);
#pragma unroll
    for (int t = 0; t < 4; ++t) {
      v16h bf = *(const v16h*)(Bs + t * 512 + lane * 16);
      acc[t] = wmma_f16(af, bf, acc[t]);
    }
  }

  int hi = (lane >> 4) & 1, nl = lane & 15;
#pragma unroll
  for (int t = 0; t < 4; ++t) {
    int col = colBase + t * 16 + nl;
    if (col < N) {
      float bv = bias ? bias[col] : 0.f;
#pragma unroll
      for (int r = 0; r < 8; ++r) {
        int row = rowBase + wave * 16 + r + 8 * hi;
        long idx = c0 + (long)row * N + col;
        float v = acc[t][r] + bv;
        if (res) v += res[idx];
        C[idx] = v;
      }
    }
  }
}

// ---------------------------------------------------------------------------
// Fused GEGLU GEMM: U = (A@W+b)[:, :Nh] * gelu((A@W+b)[:, Nh:2Nh])
// (Nh is always a multiple of 4 here -> vector B loads unconditionally)
// ---------------------------------------------------------------------------
__global__ __launch_bounds__(128) void k_gemm_geglu(const float* __restrict__ A,
                                                    const float* __restrict__ W,
                                                    const float* __restrict__ bias,
                                                    float* __restrict__ C,
                                                    int M, int Nh, int K,
                                                    long batchA, long batchC) {
  __shared__ __align__(32) half_t As[4 * 512];
  __shared__ __align__(32) half_t Ba[4 * 512];
  __shared__ __align__(32) half_t Bg[4 * 512];
  int tid = threadIdx.x, lane = tid & 31, wave = tid >> 5;
  long a0 = (long)blockIdx.z * batchA;
  long c0 = (long)blockIdx.z * batchC;
  int rowBase = blockIdx.y * 64;
  int colBase = blockIdx.x * 64;
  int N2 = 2 * Nh;
  bool aVec = (K & 3) == 0;
  v8f acca[4], accg[4];
#pragma unroll
  for (int t = 0; t < 4; ++t)
#pragma unroll
    for (int r = 0; r < 8; ++r) { acca[t][r] = 0.f; accg[t][r] = 0.f; }

  for (int k0 = 0; k0 < K; k0 += 32) {
    __syncthreads();
    load_a_tile(As, A, a0, rowBase, k0, K, aVec, tid);
    load_b_tile(Ba, W, N2, 0,  colBase, k0, K, Nh, true, tid);
    load_b_tile(Bg, W, N2, Nh, colBase, k0, K, Nh, true, tid);
    __syncthreads();
    v16h af = *(const v16h*)(As + wave * 512 + lane * 16);
#pragma unroll
    for (int t = 0; t < 4; ++t) {
      v16h bfa = *(const v16h*)(Ba + t * 512 + lane * 16);
      acca[t] = wmma_f16(af, bfa, acca[t]);
      v16h bfg = *(const v16h*)(Bg + t * 512 + lane * 16);
      accg[t] = wmma_f16(af, bfg, accg[t]);
    }
  }

  int hi = (lane >> 4) & 1, nl = lane & 15;
#pragma unroll
  for (int t = 0; t < 4; ++t) {
    int col = colBase + t * 16 + nl;
    if (col < Nh) {
      float ba = bias[col], bg = bias[Nh + col];
#pragma unroll
      for (int r = 0; r < 8; ++r) {
        int row = rowBase + wave * 16 + r + 8 * hi;
        float a = acca[t][r] + ba;
        float g = accg[t][r] + bg;
        float gl = 0.5f * g * (1.0f + erff(g * 0.70710678118654752f));
        C[c0 + (long)row * Nh + col] = a * gl;
      }
    }
  }
}

// ---------------------------------------------------------------------------
// WMMA flash attention: nk=256 keys, dh=64. Block = 128 query rows, 8 waves
// (one 16-row tile per wave). K/V streamed in two 128-key LDS halves, stored
// directly in B-fragment order; softmaxed P written in A-fragment order.
// headStride is a column offset (head*64) applied to Q/K/V/O bases.
// ---------------------------------------------------------------------------
__global__ __launch_bounds__(256) void k_attention(const float* __restrict__ Q,
                                                   const float* __restrict__ Kp,
                                                   const float* __restrict__ Vp,
                                                   float* __restrict__ O,
                                                   int qs, long qBatch,
                                                   int kvs, long kvBatch,
                                                   int os, long oBatch,
                                                   int headStride, float scale) {
  // K^T fragments: [n_tile(8)][d_half(2)] ; V fragments: [key_blk(4)][d_tile(4)]
  __shared__ __align__(32) half_t Ksw[16 * 512];
  __shared__ __align__(32) half_t Vsw[16 * 512];
  __shared__ __align__(32) half_t Psw[8 * 512];   // per-wave P A-fragment

  int tid = threadIdx.x, lane = tid & 31, wave = tid >> 5;
  int hi = (lane >> 4) & 1, nl = lane & 15;
  int h = blockIdx.y;
  const float* baseQ = Q + (long)blockIdx.z * qBatch + (long)h * headStride;
  const float* baseK = Kp + (long)blockIdx.z * kvBatch + (long)h * headStride;
  const float* baseV = Vp + (long)blockIdx.z * kvBatch + (long)h * headStride;
  float* baseO = O + (long)blockIdx.z * oBatch + (long)h * headStride;
  int q0 = blockIdx.x * 128 + wave * 16;

  // Q fragments (dh=64 -> two 32-wide k-steps), float4 reads (8-half groups
  // are contiguous in global once the +8*hi lane offset is applied).
  v16h qf[2];
  {
    const float* qp = baseQ + (long)(q0 + nl) * qs + 8 * hi;
#pragma unroll
    for (int s = 0; s < 2; ++s)
#pragma unroll
      for (int g = 0; g < 2; ++g) {
        float4 u0 = *(const float4*)(qp + 32 * s + 16 * g);
        float4 u1 = *(const float4*)(qp + 32 * s + 16 * g + 4);
        qf[s][8 * g + 0] = (half_t)u0.x; qf[s][8 * g + 1] = (half_t)u0.y;
        qf[s][8 * g + 2] = (half_t)u0.z; qf[s][8 * g + 3] = (half_t)u0.w;
        qf[s][8 * g + 4] = (half_t)u1.x; qf[s][8 * g + 5] = (half_t)u1.y;
        qf[s][8 * g + 6] = (half_t)u1.z; qf[s][8 * g + 7] = (half_t)u1.w;
      }
  }

  v8f oacc[4];
  float mrow[8], srow[8];
#pragma unroll
  for (int t = 0; t < 4; ++t)
#pragma unroll
    for (int r = 0; r < 8; ++r) oacc[t][r] = 0.f;
#pragma unroll
  for (int r = 0; r < 8; ++r) { mrow[r] = -1e30f; srow[r] = 0.f; }

  for (int kb = 0; kb < 2; ++kb) {
    __syncthreads();
    for (int i = tid; i < 2048; i += 256) {   // 128 keys x 64 dh, float4
      int n = i >> 4, d4 = (i & 15) << 2;
      long roff = (long)(kb * 128 + n) * kvs + d4;
      float4 kv4 = *(const float4*)(baseK + roff);
      float4 vv4 = *(const float4*)(baseV + roff);
      // S-GEMM B-frag: contraction = d -> 4 contiguous halfs, one b64 store
      *(v4h*)(Ksw + ((n >> 4) * 2 + (d4 >> 5)) * 512 + b_swz(d4 & 31, n & 15)) =
          cvt4(kv4);
      // PV-GEMM B-frag: contraction = key n, col = d (scattered lanes)
      int vbase = ((n >> 5) * 4 + (d4 >> 4)) * 512;
      int d16 = d4 & 15, n32 = n & 31;
      Vsw[vbase + b_swz(n32, d16 + 0)] = (half_t)vv4.x;
      Vsw[vbase + b_swz(n32, d16 + 1)] = (half_t)vv4.y;
      Vsw[vbase + b_swz(n32, d16 + 2)] = (half_t)vv4.z;
      Vsw[vbase + b_swz(n32, d16 + 3)] = (half_t)vv4.w;
    }
    __syncthreads();

    for (int kk = 0; kk < 4; ++kk) {
      v8f s0, s1;
#pragma unroll
      for (int r = 0; r < 8; ++r) { s0[r] = 0.f; s1[r] = 0.f; }
#pragma unroll
      for (int s = 0; s < 2; ++s) {
        v16h b0 = *(const v16h*)(Ksw + ((2 * kk) * 2 + s) * 512 + lane * 16);
        v16h b1 = *(const v16h*)(Ksw + ((2 * kk + 1) * 2 + s) * 512 + lane * 16);
        s0 = wmma_f16(qf[s], b0, s0);
        s1 = wmma_f16(qf[s], b1, s1);
      }
      float f[8];
#pragma unroll
      for (int r = 0; r < 8; ++r) {
        s0[r] *= scale; s1[r] *= scale;
        float tm = half_max16(fmaxf(s0[r], s1[r]));
        float mn = fmaxf(mrow[r], tm);
        f[r] = __expf(mrow[r] - mn);
        mrow[r] = mn;
        s0[r] = __expf(s0[r] - mn);
        s1[r] = __expf(s1[r] - mn);
        float rs = half_sum16(s0[r] + s1[r]);
        srow[r] = srow[r] * f[r] + rs;
      }
#pragma unroll
      for (int t = 0; t < 4; ++t)
#pragma unroll
        for (int r = 0; r < 8; ++r) oacc[t][r] *= f[r];

      __syncthreads();  // previous Psw reads finished (uniform loops)
      half_t* pw = Psw + wave * 512;
#pragma unroll
      for (int r = 0; r < 8; ++r) {
        int m = r + 8 * hi;   // D-layout row this lane owns
        pw[a_swz(m, nl)]      = (half_t)s0[r];
        pw[a_swz(m, nl + 16)] = (half_t)s1[r];
      }
      __syncthreads();
      v16h af = *(const v16h*)(Psw + wave * 512 + lane * 16);
#pragma unroll
      for (int t = 0; t < 4; ++t) {
        v16h bf = *(const v16h*)(Vsw + (kk * 4 + t) * 512 + lane * 16);
        oacc[t] = wmma_f16(af, bf, oacc[t]);
      }
    }
  }

#pragma unroll
  for (int r = 0; r < 8; ++r) {
    float inv = 1.f / srow[r];
    int row = q0 + r + 8 * hi;
#pragma unroll
    for (int t = 0; t < 4; ++t)
      baseO[(long)row * os + t * 16 + nl] = oacc[t][r] * inv;
  }
}

// ---------------------------------------------------------------------------
__global__ __launch_bounds__(256) void k_extract(const float* __restrict__ data,
                                                 float* __restrict__ out) {
  size_t p = (size_t)blockIdx.x * 256 + threadIdx.x;
  if (p >= (size_t)NB * HW_) return;
  out[p * 3 + 0] = data[p * QDIM + 0];
  out[p * 3 + 1] = data[p * QDIM + 1];
  out[p * 3 + 2] = data[p * QDIM + 2];
}

// ---------------------------------------------------------------------------
extern "C" void kernel_launch(void* const* d_in, const int* in_sizes, int n_in,
                              void* d_out, int out_size, void* d_ws,
                              size_t ws_size, hipStream_t stream) {
  (void)in_sizes; (void)n_in; (void)out_size; (void)ws_size;
  const float* x        = (const float*)d_in[0];
  const float* latents  = (const float*)d_in[1];
  const float* W_l2l    = (const float*)d_in[2];
  const float* b_l2l    = (const float*)d_in[3];
  const float* ca_ln_qg = (const float*)d_in[4];
  const float* ca_ln_qb = (const float*)d_in[5];
  const float* ca_ln_cg = (const float*)d_in[6];
  const float* ca_ln_cb = (const float*)d_in[7];
  const float* ca_Wq    = (const float*)d_in[8];
  const float* ca_Wkv   = (const float*)d_in[9];
  const float* ca_Wo    = (const float*)d_in[10];
  const float* ca_bo    = (const float*)d_in[11];
  const float* cf_ln_g  = (const float*)d_in[12];
  const float* cf_ln_b  = (const float*)d_in[13];
  const float* cf_W1    = (const float*)d_in[14];
  const float* cf_b1    = (const float*)d_in[15];
  const float* cf_W2    = (const float*)d_in[16];
  const float* cf_b2    = (const float*)d_in[17];
  const float* la_ln_g  = (const float*)d_in[18];
  const float* la_ln_b  = (const float*)d_in[19];
  const float* la_Wq    = (const float*)d_in[20];
  const float* la_Wkv   = (const float*)d_in[21];
  const float* la_Wo    = (const float*)d_in[22];
  const float* la_bo    = (const float*)d_in[23];
  const float* lf_ln_g  = (const float*)d_in[24];
  const float* lf_ln_b  = (const float*)d_in[25];
  const float* lf_W1    = (const float*)d_in[26];
  const float* lf_b1    = (const float*)d_in[27];
  const float* lf_W2    = (const float*)d_in[28];
  const float* lf_b2    = (const float*)d_in[29];

  // workspace layout (floats)
  float* ws = (float*)d_ws;
  size_t off = 0;
  float* lat  = ws + off; off += (size_t)NB * NLAT * DLAT;      // 1.05M
  float* data = ws + off; off += (size_t)NB * HW_ * QDIM;       // 3.80M
  float* tln  = ws + off; off += (size_t)NB * HW_ * QDIM;       // 3.80M
  float* cn   = ws + off; off += (size_t)NB * NLAT * DLAT;      // 1.05M
  float* bufQ = ws + off; off += (size_t)NB * HW_ * 116;        // 15.2M (q + geglu-u alias)
  float* bufU = bufQ;                                           //   live ranges disjoint
  float* bufKV = ws + off; off += (size_t)NB * NLAT * 1024;     // 2.10M
  float* bufO = ws + off; off += (size_t)NB * HW_ * 64;         // 8.39M

  k_l2l<<<(NLAT * DLAT) / 256, 256, 0, stream>>>(x, W_l2l, b_l2l, latents, lat);
  k_init_data<<<HW_ / 256, 256, 0, stream>>>(data);

  for (int i = 0; i < 4; ++i) {
    long wQ = (long)i * DLAT * DLAT, wKV = (long)i * DLAT * 1024;

    // ---- latent self-attention ----
    k_layernorm<<<(NB * NLAT) / 8, 256, 0, stream>>>(lat, tln, la_ln_g + i * DLAT,
                                                     la_ln_b + i * DLAT, DLAT);
    k_gemm<<<dim3(8, 4, NB), 128, 0, stream>>>(tln, la_Wq + wQ, nullptr, nullptr,
        bufQ, NLAT, DLAT, DLAT, (long)NLAT * DLAT, (long)NLAT * DLAT);
    k_gemm<<<dim3(16, 4, NB), 128, 0, stream>>>(tln, la_Wkv + wKV, nullptr, nullptr,
        bufKV, NLAT, 1024, DLAT, (long)NLAT * DLAT, (long)NLAT * 1024);
    k_attention<<<dim3(2, 8, NB), 256, 0, stream>>>(bufQ, bufKV, bufKV + DLAT, bufO,
        DLAT, (long)NLAT * DLAT, 1024, (long)NLAT * 1024,
        DLAT, (long)NLAT * DLAT, 64, 0.125f);
    k_gemm<<<dim3(8, 4, NB), 128, 0, stream>>>(bufO, la_Wo + wQ, la_bo + i * DLAT,
        lat, lat, NLAT, DLAT, DLAT, (long)NLAT * DLAT, (long)NLAT * DLAT);

    // ---- latent GEGLU FF ----
    k_layernorm<<<(NB * NLAT) / 8, 256, 0, stream>>>(lat, tln, lf_ln_g + i * DLAT,
                                                     lf_ln_b + i * DLAT, DLAT);
    k_gemm_geglu<<<dim3(32, 4, NB), 128, 0, stream>>>(tln,
        lf_W1 + (long)i * DLAT * 4096, lf_b1 + (long)i * 4096, bufU,
        NLAT, 2048, DLAT, (long)NLAT * DLAT, (long)NLAT * 2048);
    k_gemm<<<dim3(8, 4, NB), 128, 0, stream>>>(bufU, lf_W2 + (long)i * 2048 * DLAT,
        lf_b2 + i * DLAT, lat, lat, NLAT, DLAT, 2048,
        (long)NLAT * 2048, (long)NLAT * DLAT);

    // ---- cross attention (pixels query latents) ----
    k_layernorm<<<(NB * HW_) / 8, 256, 0, stream>>>(data, tln, ca_ln_qg + i * QDIM,
                                                    ca_ln_qb + i * QDIM, QDIM);
    k_layernorm<<<(NB * NLAT) / 8, 256, 0, stream>>>(lat, cn, ca_ln_cg + i * DLAT,
                                                     ca_ln_cb + i * DLAT, DLAT);
    k_gemm<<<dim3(1, HW_ / 64, NB), 128, 0, stream>>>(tln, ca_Wq + (long)i * QDIM * 64,
        nullptr, nullptr, bufQ, HW_, 64, QDIM, (long)HW_ * QDIM, (long)HW_ * 64);
    k_gemm<<<dim3(2, 4, NB), 128, 0, stream>>>(cn, ca_Wkv + (long)i * DLAT * 128,
        nullptr, nullptr, bufKV, NLAT, 128, DLAT, (long)NLAT * DLAT, (long)NLAT * 128);
    k_attention<<<dim3(HW_ / 128, 1, NB), 256, 0, stream>>>(bufQ, bufKV, bufKV + 64,
        bufO, 64, (long)HW_ * 64, 128, (long)NLAT * 128,
        64, (long)HW_ * 64, 0, 0.125f);
    k_gemm<<<dim3(1, HW_ / 64, NB), 128, 0, stream>>>(bufO, ca_Wo + (long)i * 64 * QDIM,
        ca_bo + i * QDIM, data, data, HW_, QDIM, 64,
        (long)HW_ * 64, (long)HW_ * QDIM);

    // ---- cross GEGLU FF ----
    k_layernorm<<<(NB * HW_) / 8, 256, 0, stream>>>(data, tln, cf_ln_g + i * QDIM,
                                                    cf_ln_b + i * QDIM, QDIM);
    k_gemm_geglu<<<dim3(2, HW_ / 64, NB), 128, 0, stream>>>(tln,
        cf_W1 + (long)i * QDIM * 232, cf_b1 + (long)i * 232, bufU,
        HW_, 116, QDIM, (long)HW_ * QDIM, (long)HW_ * 116);
    k_gemm<<<dim3(1, HW_ / 64, NB), 128, 0, stream>>>(bufU,
        cf_W2 + (long)i * 116 * QDIM, cf_b2 + i * QDIM, data, data,
        HW_, QDIM, 116, (long)HW_ * 116, (long)HW_ * QDIM);
  }

  k_extract<<<(NB * HW_) / 256, 256, 0, stream>>>(data, (float*)d_out);
}
